// SumInd_50500225467015
// MI455X (gfx1250) — matrix operand order repaired and checked by reading, not compiled, not run on hardware
//
#include <hip/hip_runtime.h>
#include <math.h>

typedef __attribute__((ext_vector_type(8)))  __bf16 bf16x8;
typedef __attribute__((ext_vector_type(16))) __bf16 bf16x16;
typedef __attribute__((ext_vector_type(8)))  float  f32x8;
typedef __attribute__((ext_vector_type(4)))  float  f32x4;

#define B_    32
#define T_    2048
#define IN_   1024
#define HID_  128
#define F_    256          // 2*HID
#define BT_   (B_ * T_)    // 65536
#define EPS_  1e-5f

// ---------------------------------------------------------------------------
// fp32 -> bf16 conversion of x (vectorized 4-wide)
// ---------------------------------------------------------------------------
__global__ void cvt_f32_to_bf16_x4(const float* __restrict__ src,
                                   __bf16* __restrict__ dst, long long n4) {
  long long i = (long long)blockIdx.x * blockDim.x + threadIdx.x;
  if (i >= n4) return;
  f32x4 v = *(const f32x4*)(src + i * 4);
  __bf16 o0 = (__bf16)v.x, o1 = (__bf16)v.y, o2 = (__bf16)v.z, o3 = (__bf16)v.w;
  dst[i * 4 + 0] = o0; dst[i * 4 + 1] = o1;
  dst[i * 4 + 2] = o2; dst[i * 4 + 3] = o3;
}

// Pack forward + backward weight matrices [HID,K] each into one bf16 [F_,K]
__global__ void pack2_w_bf16(const float* __restrict__ Wf,
                             const float* __restrict__ Wb,
                             __bf16* __restrict__ out, int K) {
  int i = blockIdx.x * blockDim.x + threadIdx.x;  // over F_*K
  if (i >= F_ * K) return;
  int row = i / K, col = i - row * K;
  float v = (row < HID_) ? Wf[row * K + col] : Wb[(row - HID_) * K + col];
  out[i] = (__bf16)v;
}

__global__ void zero_f32(float* __restrict__ p, int n) {
  int i = blockIdx.x * blockDim.x + threadIdx.x;
  if (i < n) p[i] = 0.0f;
}

// ---------------------------------------------------------------------------
// bf16 WMMA GEMM:  C[M,N] = A[M,K] * W[N,K]^T   (A,W bf16 row-major, C fp32)
// One wave computes a 16x64 C strip (4 tiles): A fragment reused 4x.
// 8 waves / block -> 128 rows x 64 cols per block.
// ---------------------------------------------------------------------------
__global__ void gemm_bf16_wmma(const __bf16* __restrict__ A,
                               const __bf16* __restrict__ W,
                               float* __restrict__ C,
                               int M, int N, int K) {
  const int wave  = threadIdx.x >> 5;
  const int lane  = threadIdx.x & 31;
  const int m0    = (blockIdx.x * 8 + wave) * 16;
  const int n0    = blockIdx.y * 64;
  const int l16   = lane & 15;
  const int half  = lane >> 4;

  const __bf16* aRow = A + (size_t)(m0 + l16) * K;
  const __bf16* bRow0 = W + (size_t)(n0 +  0 + l16) * K;
  const __bf16* bRow1 = W + (size_t)(n0 + 16 + l16) * K;
  const __bf16* bRow2 = W + (size_t)(n0 + 32 + l16) * K;
  const __bf16* bRow3 = W + (size_t)(n0 + 48 + l16) * K;

  f32x8 acc0 = {}, acc1 = {}, acc2 = {}, acc3 = {};
  const int aOff = half * 16;   // A layout: lane-half picks K+0..15 / K+16..31
  const int bOff = half * 16;   // B layout: lane-half picks K+0..15 / K+16..31

  for (int k = 0; k < K; k += 32) {
    __builtin_prefetch(aRow + k + 256, 0, 1);   // global_prefetch_b8 on the A stream
    bf16x8 a0 = *(const bf16x8*)(aRow + k + aOff);
    bf16x8 a1 = *(const bf16x8*)(aRow + k + aOff + 8);
    bf16x16 av = __builtin_shufflevector(a0, a1, 0,1,2,3,4,5,6,7,8,9,10,11,12,13,14,15);

    bf16x8 b0lo = *(const bf16x8*)(bRow0 + k + bOff);
    bf16x8 b0hi = *(const bf16x8*)(bRow0 + k + bOff + 8);
    bf16x8 b1lo = *(const bf16x8*)(bRow1 + k + bOff);
    bf16x8 b1hi = *(const bf16x8*)(bRow1 + k + bOff + 8);
    bf16x8 b2lo = *(const bf16x8*)(bRow2 + k + bOff);
    bf16x8 b2hi = *(const bf16x8*)(bRow2 + k + bOff + 8);
    bf16x8 b3lo = *(const bf16x8*)(bRow3 + k + bOff);
    bf16x8 b3hi = *(const bf16x8*)(bRow3 + k + bOff + 8);

    bf16x16 bv0 = __builtin_shufflevector(b0lo, b0hi, 0,1,2,3,4,5,6,7,8,9,10,11,12,13,14,15);
    bf16x16 bv1 = __builtin_shufflevector(b1lo, b1hi, 0,1,2,3,4,5,6,7,8,9,10,11,12,13,14,15);
    bf16x16 bv2 = __builtin_shufflevector(b2lo, b2hi, 0,1,2,3,4,5,6,7,8,9,10,11,12,13,14,15);
    bf16x16 bv3 = __builtin_shufflevector(b3lo, b3hi, 0,1,2,3,4,5,6,7,8,9,10,11,12,13,14,15);

    acc0 = __builtin_amdgcn_wmma_f32_16x16x32_bf16(false, av, false, bv0, (short)0, acc0, false, false);
    acc1 = __builtin_amdgcn_wmma_f32_16x16x32_bf16(false, av, false, bv1, (short)0, acc1, false, false);
    acc2 = __builtin_amdgcn_wmma_f32_16x16x32_bf16(false, av, false, bv2, (short)0, acc2, false, false);
    acc3 = __builtin_amdgcn_wmma_f32_16x16x32_bf16(false, av, false, bv3, (short)0, acc3, false, false);
  }

  // C/D layout: VGPR r holds (M = m0 + r + 8*half, N = ntile + l16)
  const int mBase = m0 + half * 8;
  float* Cr = C + (size_t)mBase * N + n0 + l16;
#pragma unroll
  for (int r = 0; r < 8; ++r) {
    Cr[(size_t)r * N +  0] = acc0[r];
    Cr[(size_t)r * N + 16] = acc1[r];
    Cr[(size_t)r * N + 32] = acc2[r];
    Cr[(size_t)r * N + 48] = acc3[r];
  }
}

// ---------------------------------------------------------------------------
// IndRNN scan: h_t = relu(p_t + bias + u * h_{t+/-1}), both directions.
// Block = one batch row, 256 threads = 256 features (first 128 fwd, last bwd).
// Accumulates per-feature sum / sum-of-squares for BatchNorm.
// Output type templated: layer0 emits bf16 (feeds next GEMM), layer1 fp32.
// ---------------------------------------------------------------------------
template <typename OutT>
__global__ void indrnn_scan(const float* __restrict__ proj,
                            const float* __restrict__ bf, const float* __restrict__ bb,
                            const float* __restrict__ uf, const float* __restrict__ ub,
                            OutT* __restrict__ h, float* __restrict__ stats) {
  const int b = blockIdx.x;
  const int f = threadIdx.x;            // 0..255
  const bool bwd = f >= HID_;
  const int hf  = bwd ? f - HID_ : f;
  const float bias = bwd ? bb[hf] : bf[hf];
  const float u    = bwd ? ub[hf] : uf[hf];

  const float* p = proj + (size_t)b * T_ * F_ + f;
  OutT* o        = h    + (size_t)b * T_ * F_ + f;

  float hv = 0.0f, s = 0.0f, ss = 0.0f;
  if (!bwd) {
    for (int t = 0; t < T_; ++t) {
      float x = p[(size_t)t * F_];
      hv = fmaxf(0.0f, fmaf(u, hv, x + bias));
      o[(size_t)t * F_] = (OutT)hv;
      s += hv; ss = fmaf(hv, hv, ss);
    }
  } else {
    for (int t = T_ - 1; t >= 0; --t) {
      float x = p[(size_t)t * F_];
      hv = fmaxf(0.0f, fmaf(u, hv, x + bias));
      o[(size_t)t * F_] = (OutT)hv;
      s += hv; ss = fmaf(hv, hv, ss);
    }
  }
  atomicAdd(&stats[f], s);
  atomicAdd(&stats[F_ + f], ss);
}

// ---------------------------------------------------------------------------
// Fold BN0 into layer-1 weights:  (h*s + t)·W1^T == h·(W1*s)^T + t·W1^T
// One block, 256 threads; thread n owns output row n of W1fold + bias1fold[n].
// ---------------------------------------------------------------------------
__global__ void fold_bn_into_w1(const float* __restrict__ stats,
                                const float* __restrict__ g, const float* __restrict__ be,
                                const float* __restrict__ W10, const float* __restrict__ W11,
                                const float* __restrict__ b10, const float* __restrict__ b11,
                                __bf16* __restrict__ wOut, float* __restrict__ biasOut) {
  __shared__ float sc[F_], sh[F_];
  const int n = threadIdx.x;            // 0..255
  const float invN = 1.0f / (float)BT_;
  {
    float mean = stats[n] * invN;
    float var  = fmaf(stats[F_ + n], invN, -mean * mean);
    float sv   = g[n] * rsqrtf(var + EPS_);
    sc[n] = sv;
    sh[n] = be[n] - mean * sv;
  }
  __syncthreads();
  const float* Wrow = (n < HID_) ? (W10 + (size_t)n * F_) : (W11 + (size_t)(n - HID_) * F_);
  float bias = (n < HID_) ? b10[n] : b11[n - HID_];
  for (int k = 0; k < F_; ++k) {
    float w = Wrow[k];
    wOut[(size_t)n * F_ + k] = (__bf16)(w * sc[k]);
    bias = fmaf(w, sh[k], bias);
  }
  biasOut[n] = bias;
}

// ---------------------------------------------------------------------------
// Fold BN1 into head weights: sum_f lw[f]*BN(h[f]) = sum_f h[f]*wfold[f] + c
// ---------------------------------------------------------------------------
__global__ void fold_head(const float* __restrict__ stats,
                          const float* __restrict__ g, const float* __restrict__ be,
                          const float* __restrict__ lw, const float* __restrict__ lb,
                          float* __restrict__ wfold, float* __restrict__ cOut) {
  __shared__ float red[F_];
  const int f = threadIdx.x;
  const float invN = 1.0f / (float)BT_;
  float mean = stats[f] * invN;
  float var  = fmaf(stats[F_ + f], invN, -mean * mean);
  float sv   = g[f] * rsqrtf(var + EPS_);
  wfold[f] = lw[f] * sv;
  red[f] = lw[f] * (be[f] - mean * sv);
  __syncthreads();
#pragma unroll
  for (int st = 128; st > 0; st >>= 1) {
    if (f < st) red[f] += red[f + st];
    __syncthreads();
  }
  if (f == 0) cOut[0] = red[0] + lb[0];
}

// ---------------------------------------------------------------------------
// Head: folded 256-wide dot + sigmoid. One wave per output row.
// ---------------------------------------------------------------------------
__global__ void head_dot_sigmoid(const float* __restrict__ h1,
                                 const float* __restrict__ wfold,
                                 const float* __restrict__ c,
                                 float* __restrict__ out, int rows) {
  int gw   = (int)((blockIdx.x * (long long)blockDim.x + threadIdx.x) >> 5);
  int lane = threadIdx.x & 31;
  if (gw >= rows) return;
  const float* row = h1 + (size_t)gw * F_;
  float sum = 0.0f;
#pragma unroll
  for (int j = 0; j < F_ / 32; ++j) {
    int f = lane + j * 32;
    sum = fmaf(row[f], wfold[f], sum);
  }
#pragma unroll
  for (int off = 16; off > 0; off >>= 1) sum += __shfl_down(sum, off, 32);
  if (lane == 0) out[gw] = 1.0f / (1.0f + __expf(-(sum + c[0])));
}

// ---------------------------------------------------------------------------
extern "C" void kernel_launch(void* const* d_in, const int* in_sizes, int n_in,
                              void* d_out, int out_size, void* d_ws, size_t ws_size,
                              hipStream_t stream) {
  const float* x    = (const float*)d_in[0];
  const float* W00  = (const float*)d_in[1];
  const float* W01  = (const float*)d_in[2];
  const float* W10  = (const float*)d_in[3];
  const float* W11  = (const float*)d_in[4];
  const float* b00  = (const float*)d_in[5];
  const float* b01  = (const float*)d_in[6];
  const float* b10  = (const float*)d_in[7];
  const float* b11  = (const float*)d_in[8];
  const float* u00  = (const float*)d_in[9];
  const float* u01  = (const float*)d_in[10];
  const float* u10  = (const float*)d_in[11];
  const float* u11  = (const float*)d_in[12];
  const float* g0   = (const float*)d_in[13];
  const float* be0  = (const float*)d_in[14];
  const float* g1   = (const float*)d_in[15];
  const float* be1  = (const float*)d_in[16];
  const float* lw   = (const float*)d_in[17];
  const float* lb   = (const float*)d_in[18];
  float* out = (float*)d_out;

  char* ws = (char*)d_ws;
  size_t off = 0;
  auto alloc = [&](size_t bytes) { char* p = ws + off; off = (off + bytes + 255) & ~(size_t)255; return p; };

  __bf16* xb     = (__bf16*)alloc((size_t)BT_ * IN_ * 2);   // 128 MB
  __bf16* w0b    = (__bf16*)alloc((size_t)F_ * IN_ * 2);    // 512 KB
  __bf16* w1b    = (__bf16*)alloc((size_t)F_ * F_ * 2);     // 128 KB  (BN0-folded)
  float*  proj   = (float*) alloc((size_t)BT_ * F_ * 4);    // 64 MB   (reused layer1)
  __bf16* hb     = (__bf16*)alloc((size_t)BT_ * F_ * 2);    // 32 MB   (h0, bf16)
  float*  h1     = (float*) alloc((size_t)BT_ * F_ * 4);    // 64 MB
  float*  stats0 = (float*) alloc(2 * F_ * 4);
  float*  stats1 = (float*) alloc(2 * F_ * 4);
  float*  bias1  = (float*) alloc(F_ * 4);
  float*  wfold  = (float*) alloc(F_ * 4);
  float*  cbuf   = (float*) alloc(4);
  (void)ws_size; (void)out_size; (void)n_in; (void)in_sizes;

  // 0) zero BN statistics
  zero_f32<<<dim3(2), dim3(256), 0, stream>>>(stats0, 2 * F_);
  zero_f32<<<dim3(2), dim3(256), 0, stream>>>(stats1, 2 * F_);

  // 1) x -> bf16
  {
    long long n4 = (long long)BT_ * IN_ / 4;
    cvt_f32_to_bf16_x4<<<dim3((unsigned)((n4 + 255) / 256)), dim3(256), 0, stream>>>(x, xb, n4);
  }
  // 2) layer-0 weights -> packed bf16
  pack2_w_bf16<<<dim3((F_ * IN_ + 255) / 256), dim3(256), 0, stream>>>(W00, W01, w0b, IN_);

  // 3) layer-0 GEMM: proj = x * [W00;W01]^T   [65536,1024]x[1024,256]
  gemm_bf16_wmma<<<dim3(BT_ / 128, F_ / 64), dim3(256), 0, stream>>>(xb, w0b, proj, BT_, F_, IN_);

  // 4) layer-0 scan -> h0 in bf16 (feeds next GEMM directly) + BN0 stats
  indrnn_scan<__bf16><<<dim3(B_), dim3(F_), 0, stream>>>(proj, b00, b01, u00, u01, hb, stats0);

  // 5) fold BN0 into layer-1 weights + bias (tiny, one block)
  fold_bn_into_w1<<<dim3(1), dim3(F_), 0, stream>>>(stats0, g0, be0, W10, W11, b10, b11, w1b, bias1);

  // 6) layer-1 GEMM: proj = h0 * W1fold^T   [65536,256]x[256,256]
  gemm_bf16_wmma<<<dim3(BT_ / 128, F_ / 64), dim3(256), 0, stream>>>(hb, w1b, proj, BT_, F_, F_);

  // 7) layer-1 scan (bias includes folded BN0 shift) -> h1 fp32 + BN1 stats
  indrnn_scan<float><<<dim3(B_), dim3(F_), 0, stream>>>(proj, bias1, bias1 + HID_, u10, u11, h1, stats1);

  // 8) fold BN1 into head weights (tiny, one block)
  fold_head<<<dim3(1), dim3(F_), 0, stream>>>(stats1, g1, be1, lw, lb, wfold, cbuf);

  // 9) head: folded dot + sigmoid, one wave per row
  head_dot_sigmoid<<<dim3(BT_ / 8), dim3(256), 0, stream>>>(h1, wfold, cbuf, out, BT_);
}
